// MultiReDiffusion_59442347376896
// MI455X (gfx1250) — matrix operand order repaired and checked by reading, not compiled, not run on hardware
//
#include <hip/hip_runtime.h>
#include <hip/hip_bf16.h>

#define R_    4
#define J_    6
#define N_    4096
#define DIN_  128
#define DOUT_ 64

typedef __attribute__((ext_vector_type(16))) __bf16       v16bf;
typedef __attribute__((ext_vector_type(8)))  float        v8f;
typedef __attribute__((ext_vector_type(2)))  float        v2f;
typedef __attribute__((ext_vector_type(4)))  unsigned int u32x4;
typedef __attribute__((ext_vector_type(8)))  int          i32x8;
typedef __attribute__((ext_vector_type(4)))  int          i32x4;

// ---------------------------------------------------------------------------
// Kernel 0: coef[r] = sum_j theta*t ; xT_{hi,lo}[r][d][k] = split_bf16(coef * x[r][k][d])
// ---------------------------------------------------------------------------
__global__ void prep_xT_kernel(const float* __restrict__ theta,
                               const float* __restrict__ tt,
                               const float* __restrict__ x,
                               __bf16* __restrict__ xT_hi,
                               __bf16* __restrict__ xT_lo) {
    int r = blockIdx.x >> 7;       // blockIdx.x = r*128 + d
    int d = blockIdx.x & 127;
    float coef = 0.0f;
#pragma unroll
    for (int j = 0; j < J_; ++j) coef += theta[r * J_ + j] * tt[r * J_ + j];

    const float* xr = x + (size_t)r * N_ * DIN_;
    size_t outBase = ((size_t)r * DIN_ + d) * N_;
    for (int k = threadIdx.x; k < N_; k += blockDim.x) {
        float v = coef * xr[(size_t)k * DIN_ + d];
        __bf16 hi = (__bf16)v;
        float  lo = v - (float)hi;
        xT_hi[outBase + k] = hi;
        xT_lo[outBase + k] = (__bf16)lo;
    }
}

// ---------------------------------------------------------------------------
// Kernel 1: S = (coef*a@x) via split-bf16 WMMA with async-LDS B staging,
// fused fc layer via f32 WMMA, bias + PReLU -> u.
// Grid: (N/64, R), 128 threads (4 waves); wave owns a 16x128 strip of S.
// ---------------------------------------------------------------------------
__launch_bounds__(128)
__global__ void diffuse_fc_kernel(const float*  __restrict__ a,
                                  const __bf16* __restrict__ xT_hi,
                                  const __bf16* __restrict__ xT_lo,
                                  const float*  __restrict__ fc_w,
                                  const float*  __restrict__ fc_b,
                                  const float*  __restrict__ p0,
                                  float* __restrict__ u_out) {
    const int r      = blockIdx.y;
    const int nbase  = blockIdx.x * 64;
    const int tid    = threadIdx.x;
    const int lane   = tid & 31;
    const int wave   = tid >> 5;
    const int hiHalf = (lane >= 16) ? 1 : 0;

    __shared__ float lds_fc[DOUT_][DIN_ + 2];               // fc_w[r], stride 130 (= TDM pad)
    __shared__ float lds_s[4][16][DIN_ + 4];                // per-wave S strips, stride 132
    __shared__ __align__(32) __bf16 lds_b[2][2][DIN_][40];  // [buf][hi/lo][d][k], 80B rows

    // ---- fc_w[r] (64x128 fp32) -> LDS: Tensor Data Mover if available ----
    const float* fw = fc_w + (size_t)r * DOUT_ * DIN_;
#if __has_builtin(__builtin_amdgcn_tensor_load_to_lds)
    if (wave == 0) {
        unsigned long long ga = (unsigned long long)(uintptr_t)fw;
        unsigned int ldsOff = (unsigned int)(uintptr_t)&lds_fc[0][0];
        u32x4 g0;
        i32x8 g1;
        i32x4 g2 = {0, 0, 0, 0}, g3 = {0, 0, 0, 0};
        i32x8 g4 = {0, 0, 0, 0, 0, 0, 0, 0};
        g0[0] = 1u;                                           // count=1, user mode
        g0[1] = ldsOff;                                       // lds_addr
        g0[2] = (unsigned int)(ga & 0xFFFFFFFFu);             // global_addr[31:0]
        g0[3] = (unsigned int)((ga >> 32) & 0x01FFFFFFu)      // global_addr[56:32]
                | 0x80000000u;                                // type=2 (image)
        // data_size=4B(2), pad_enable, pad_interval=128dw(6), pad_amount=2dw(1)
        g1[0] = (int)((2u << 16) | (1u << 20) | (6u << 22) | (1u << 25));
        g1[1] = (int)(128u << 16);                            // tensor_dim0 = 128
        g1[2] = (int)(64u  << 16);                            // tensor_dim1 = 64
        g1[3] = (int)(128u << 16);                            // tile_dim0 = 128
        g1[4] = 64;                                           // tile_dim1 = 64
        g1[5] = 128;                                          // tensor_dim0_stride = 128
        g1[6] = 0;
        g1[7] = 0;
        __builtin_amdgcn_tensor_load_to_lds(g0, g1, g2, g3, g4, 0);
        __builtin_amdgcn_s_wait_tensorcnt((short)0);
    }
#else
    for (int i = tid; i < DOUT_ * DIN_; i += blockDim.x)
        lds_fc[i >> 7][i & 127] = fw[i];
#endif

    // ---- Stage 1 setup ----
    const int rowStrip = nbase + wave * 16;
    const int aRow     = rowStrip + (lane & 15);
    const float* aBase = a + ((size_t)r * N_ + aRow) * N_;
    const __bf16* bHiBase = xT_hi + (size_t)r * DIN_ * N_;
    const __bf16* bLoBase = xT_lo + (size_t)r * DIN_ * N_;
    const int kShiftA = hiHalf ? 8 : 0;    // 16-bit A 16x32 lane layout
    const int kShiftB = hiHalf ? 8 : 0;    // 16-bit B 32x16 lane layout
    const int bCol    = lane & 15;

    // Issue the unique 16KB B tile (hi+lo) for k-chunk kbn into lds_b[buf]
    // via the async DMA path: 8 x global_load_async_to_lds_b128 per wave.
    auto issueB = [&](int buf, int kbn) {
#pragma unroll
        for (int i = 0; i < 8; ++i) {
            int c  = i * 128 + tid;            // 1024 x 16B chunks
            int h  = c >> 9;                   // 0 = hi, 1 = lo
            int dd = (c >> 2) & 127;           // column d
            int kc = c & 3;                    // 16B chunk within 64B row
            const __bf16* gp = (h ? bLoBase : bHiBase) +
                               (size_t)dd * N_ + kbn + kc * 8;
            unsigned int la = (unsigned int)(uintptr_t)&lds_b[buf][h][dd][kc * 8];
            unsigned long long gaddr = (unsigned long long)(uintptr_t)gp;
            asm volatile("global_load_async_to_lds_b128 %0, %1, off"
                         :: "v"(la), "v"(gaddr) : "memory");
        }
    };

    v8f acc[8] = {};
    issueB(0, 0);
    int cur = 0;

    for (int kb = 0; kb < N_; kb += 32) {
        // Prefetch next buffer while computing on the current one.
        issueB(cur ^ 1, (kb + 32) & (N_ - 1));
        // Async loads complete in order: <=8 outstanding => current buffer done.
        asm volatile("s_wait_asynccnt 0x8" ::: "memory");
        __syncthreads();

        // Prefetch upcoming A stream (global_prefetch_b8).
        __builtin_prefetch(aBase + ((kb + 64) & (N_ - 1)), 0, 1);

        // A fragment: two runs of 8 consecutive K values per lane, split hi/lo.
        const int k0 = kb + kShiftA;
        alignas(16) float av[16];
        *(float4*)&av[0]  = *(const float4*)(aBase + k0);
        *(float4*)&av[4]  = *(const float4*)(aBase + k0 + 4);
        *(float4*)&av[8]  = *(const float4*)(aBase + k0 + 16);
        *(float4*)&av[12] = *(const float4*)(aBase + k0 + 20);
        v16bf ahi, alo;
#pragma unroll
        for (int e = 0; e < 16; ++e) {
            __bf16 h = (__bf16)av[e];
            ahi[e] = h;
            alo[e] = (__bf16)(av[e] - (float)h);
        }

#pragma unroll
        for (int t = 0; t < 8; ++t) {
            const __bf16* bp = &lds_b[cur][0][t * 16 + bCol][kShiftB];
            const __bf16* lp = &lds_b[cur][1][t * 16 + bCol][kShiftB];
            union Frag { uint4 q[2]; v16bf v; };
            Frag bh, bl;
            bh.q[0] = *(const uint4*)(bp);
            bh.q[1] = *(const uint4*)(bp + 16);
            bl.q[0] = *(const uint4*)(lp);
            bl.q[1] = *(const uint4*)(lp + 16);
            acc[t] = __builtin_amdgcn_wmma_f32_16x16x32_bf16(
                false, ahi, false, bh.v, (short)0, acc[t], false, false);
            acc[t] = __builtin_amdgcn_wmma_f32_16x16x32_bf16(
                false, ahi, false, bl.v, (short)0, acc[t], false, false);
            acc[t] = __builtin_amdgcn_wmma_f32_16x16x32_bf16(
                false, alo, false, bh.v, (short)0, acc[t], false, false);
        }
        __syncthreads();   // done reading lds_b[cur] before it is refilled
        cur ^= 1;
    }

    // Spill S strip to LDS in C-matrix layout (VGPR j: M=j(+8), N=lane&15).
    const int cRowOff = hiHalf ? 8 : 0;
#pragma unroll
    for (int t = 0; t < 8; ++t)
#pragma unroll
        for (int j = 0; j < 8; ++j)
            lds_s[wave][j + cRowOff][t * 16 + bCol] = acc[t][j];

    __syncthreads();   // fc_w LDS + own strip visible

    // ---- Stage 2: u strip = prelu(S_strip @ fc_w^T + fc_b)  (exact f32 WMMA) ----
    const int aRow2 = lane & 15;
    const int aKoff = hiHalf ? 2 : 0;   // f32 A 16x4 lane layout
    const int bKoff = hiHalf ? 1 : 0;   // f32 B 4x16 lane layout
    v8f uacc[4] = {};

    for (int kk = 0; kk < DIN_; kk += 4) {
        v2f a2;
        a2[0] = lds_s[wave][aRow2][kk + aKoff + 0];
        a2[1] = lds_s[wave][aRow2][kk + aKoff + 1];
#pragma unroll
        for (int ot = 0; ot < 4; ++ot) {
            v2f b2;
            b2[0] = lds_fc[ot * 16 + bCol][kk + bKoff + 0];
            b2[1] = lds_fc[ot * 16 + bCol][kk + bKoff + 2];
            uacc[ot] = __builtin_amdgcn_wmma_f32_16x16x4_f32(
                false, a2, false, b2, (short)0, uacc[ot], false, false);
        }
    }

    const float p = p0[0];
    const float* fb = fc_b + r * DOUT_;
    float* uO = u_out + (size_t)r * N_ * DOUT_;
#pragma unroll
    for (int ot = 0; ot < 4; ++ot) {
        const int o = ot * 16 + bCol;
        const float bias = fb[o];
#pragma unroll
        for (int j = 0; j < 8; ++j) {
            const int row = rowStrip + j + cRowOff;
            float v = uacc[ot][j] + bias;
            v = (v >= 0.0f) ? v : p * v;
            uO[(size_t)row * DOUT_ + o] = v;
        }
    }
}

// ---------------------------------------------------------------------------
// Kernel 2: h[q,n,o] = prelu(sum_r conv_w[q,r]*u[r,n,o] + conv_b[q], p1)
// ---------------------------------------------------------------------------
__global__ void mix_kernel(const float* __restrict__ u,
                           const float* __restrict__ conv_w,
                           const float* __restrict__ conv_b,
                           const float* __restrict__ p1,
                           float* __restrict__ h) {
    size_t idx = (size_t)blockIdx.x * blockDim.x + threadIdx.x;
    if (idx >= (size_t)N_ * DOUT_) return;
    float uv[R_];
#pragma unroll
    for (int rr = 0; rr < R_; ++rr) uv[rr] = u[(size_t)rr * N_ * DOUT_ + idx];
    const float p = p1[0];
#pragma unroll
    for (int q = 0; q < R_; ++q) {
        float s = conv_b[q];
#pragma unroll
        for (int rr = 0; rr < R_; ++rr) s += conv_w[q * R_ + rr] * uv[rr];
        h[(size_t)q * N_ * DOUT_ + idx] = (s >= 0.0f) ? s : p * s;
    }
}

// ---------------------------------------------------------------------------
extern "C" void kernel_launch(void* const* d_in, const int* in_sizes, int n_in,
                              void* d_out, int out_size, void* d_ws, size_t ws_size,
                              hipStream_t stream) {
    (void)in_sizes; (void)n_in; (void)out_size; (void)ws_size;
    const float* theta  = (const float*)d_in[0];
    const float* tt     = (const float*)d_in[1];
    const float* a      = (const float*)d_in[2];
    const float* x      = (const float*)d_in[3];
    const float* fc_w   = (const float*)d_in[4];
    const float* fc_b   = (const float*)d_in[5];
    const float* conv_w = (const float*)d_in[6];
    const float* conv_b = (const float*)d_in[7];
    const float* p0     = (const float*)d_in[8];
    const float* p1     = (const float*)d_in[9];

    float* h_out = (float*)d_out;                              // first output
    float* u_out = h_out + (size_t)R_ * N_ * DOUT_;            // second output

    __bf16* xT_hi = (__bf16*)d_ws;                             // 4 MB
    __bf16* xT_lo = xT_hi + (size_t)R_ * DIN_ * N_;            // 4 MB

    prep_xT_kernel<<<R_ * DIN_, 256, 0, stream>>>(theta, tt, x, xT_hi, xT_lo);

    dim3 grid(N_ / 64, R_);
    diffuse_fc_kernel<<<grid, 128, 0, stream>>>(a, xT_hi, xT_lo, fc_w, fc_b, p0, u_out);

    int total = N_ * DOUT_;
    mix_kernel<<<(total + 255) / 256, 256, 0, stream>>>(u_out, conv_w, conv_b, p1, h_out);
}
